// FlashSVDRoPEAttention_37812892074289
// MI455X (gfx1250) — compile-verified, matrix operand order checked by reading, not run on hardware
//
#include <hip/hip_runtime.h>
#include <hip/hip_bf16.h>

typedef _Float16 half_t;
typedef __attribute__((ext_vector_type(16))) _Float16 v16h;
typedef __attribute__((ext_vector_type(8)))  _Float16 v8h;
typedef __attribute__((ext_vector_type(8)))  float    v8f;

union Frag { v16h v; v8h h[2]; };

constexpr int kH  = 16;    // heads
constexpr int kDH = 64;    // head dim
constexpr int kR  = 256;   // SVD rank
constexpr int kM  = 1024;  // sequence length
constexpr int kD  = 1024;  // model dim = kH*kDH
// ln(10000)/32
constexpr float kLogThetaOverHalf = 0.2878231366242558f;

__device__ __forceinline__ v8f wmma_f16(const Frag& a, const Frag& b, v8f c) {
  // D = A(16x32 f16) * B(32x16 f16) + C(16x16 f32)
  return __builtin_amdgcn_wmma_f32_16x16x32_f16(false, a.v, false, b.v,
                                                (short)0, c, false, false);
}

// Async 16B/lane copy global -> LDS, tracked by ASYNCcnt.
// lds_off: byte offset within LDS; gl_off: byte offset from saddr base.
__device__ __forceinline__ void async_copy_b128(unsigned lds_off, unsigned gl_off,
                                                const half_t* base) {
  asm volatile("global_load_async_to_lds_b128 %0, %1, %2"
               :: "v"(lds_off), "v"(gl_off), "s"(base) : "memory");
}

// ---------------------------------------------------------------------------
// Kernel A: fused SVD projection + bias + RoPE, fp32 -> f16 workspace.
//   Q,K  -> [B][H][M][DH] row-major f16 (Q pre-scaled by 1/sqrt(DH))
//   V    -> [B][H][DH][M] (transposed) f16, so kernel B's V B-fragments are
//           contiguous 16B loads.
// Block = 128 threads (4 waves). Each block: 64(M) x 64(dh of one head).
// ---------------------------------------------------------------------------
__global__ __launch_bounds__(128)
void svd_proj_rope_kernel(const float* __restrict__ Pq, const float* __restrict__ Pk,
                          const float* __restrict__ Pv,
                          const float* __restrict__ Vq, const float* __restrict__ Vk,
                          const float* __restrict__ Vv,
                          const float* __restrict__ bq, const float* __restrict__ bk,
                          const float* __restrict__ bv,
                          const int*   __restrict__ pos,
                          half_t* __restrict__ Qh, half_t* __restrict__ Kh,
                          half_t* __restrict__ Vth)
{
  const int mt = blockIdx.x;                  // M tile (64 rows)
  const int hh = blockIdx.y;                  // head (= 64-wide column tile of d)
  const int b  = blockIdx.z / 3;
  const int which = blockIdx.z % 3;           // 0=q, 1=k, 2=v
  const int tid  = threadIdx.x;
  const int lane = tid & 31;
  const int wave = tid >> 5;

  const float* P    = (which == 0) ? Pq : (which == 1) ? Pk : Pv;
  const float* Vw   = (which == 0) ? Vq : (which == 1) ? Vk : Vv;
  const float* bias = (which == 0) ? bq : (which == 1) ? bk : bv;

  // LDS: A tile [64 m][40 k] f16, B tile transposed [64 n][40 k] f16.
  __shared__ half_t Asub[64 * 40];
  __shared__ half_t Bsub[64 * 40];
  __shared__ half_t Vtr[64 * 72];             // [dh][m] for coalesced V store

  v8f acc[4] = {};                            // 16 rows x 64 cols per wave

  const int col  = lane & 15;
  const int rsel = (lane >> 4) * 8;           // C-frag: lanes>=16 hold rows +8
  const int koff = (lane < 16) ? 0 : 8;       // A/B-frag K sub-offset

  for (int ks = 0; ks < kR; ks += 32) {
    // --- stage A tile (64x32 fp32 -> f16) ---
    {
      const int r  = tid >> 1;
      const int c0 = (tid & 1) * 16;
      const float4* src =
          (const float4*)(P + ((size_t)b * kM + (size_t)mt * 64 + r) * kR + ks + c0);
      #pragma unroll
      for (int i = 0; i < 4; ++i) {
        float4 f = src[i];
        half_t* dst = &Asub[r * 40 + c0 + i * 4];
        dst[0] = (half_t)f.x; dst[1] = (half_t)f.y;
        dst[2] = (half_t)f.z; dst[3] = (half_t)f.w;
      }
    }
    // --- stage B tile (32x64 fp32, store transposed [n][k] f16) ---
    {
      const int k  = tid >> 2;
      const int c0 = (tid & 3) * 16;
      const float4* src =
          (const float4*)(Vw + (size_t)(ks + k) * kD + hh * 64 + c0);
      #pragma unroll
      for (int i = 0; i < 4; ++i) {
        float4 f = src[i];
        const int n = c0 + i * 4;
        Bsub[(n + 0) * 40 + k] = (half_t)f.x;
        Bsub[(n + 1) * 40 + k] = (half_t)f.y;
        Bsub[(n + 2) * 40 + k] = (half_t)f.z;
        Bsub[(n + 3) * 40 + k] = (half_t)f.w;
      }
    }
    __syncthreads();

    // --- WMMA: each wave does 16 rows x 4 n-subtiles ---
    const int r = wave * 16 + col;
    Frag a;
    a.h[0] = *(const v8h*)&Asub[r * 40 + koff];
    a.h[1] = *(const v8h*)&Asub[r * 40 + koff + 16];
    #pragma unroll
    for (int t = 0; t < 4; ++t) {
      const int n = t * 16 + col;
      Frag bb;
      bb.h[0] = *(const v8h*)&Bsub[n * 40 + koff];
      bb.h[1] = *(const v8h*)&Bsub[n * 40 + koff + 16];
      acc[t] = wmma_f16(a, bb, acc[t]);
    }
    __syncthreads();
  }

  // --- bias ---
  #pragma unroll
  for (int t = 0; t < 4; ++t) {
    const float bval = bias[hh * 64 + t * 16 + col];
    #pragma unroll
    for (int j = 0; j < 8; ++j) acc[t][j] += bval;
  }

  if (which < 2) {
    // --- RoPE in registers: frag t pairs with frag t+2 (dh <-> dh+32) ---
    #pragma unroll
    for (int t = 0; t < 2; ++t) {
      const float invf = __expf(-(float)(t * 16 + col) * kLogThetaOverHalf);
      #pragma unroll
      for (int j = 0; j < 8; ++j) {
        const int m = mt * 64 + wave * 16 + rsel + j;
        const float ang = (float)pos[(size_t)b * kM + m] * invf;
        float s, c;
        __sincosf(ang, &s, &c);
        const float x0 = acc[t][j], x1 = acc[t + 2][j];
        acc[t][j]     = x0 * c - x1 * s;
        acc[t + 2][j] = x0 * s + x1 * c;
      }
    }
    const float qscale = (which == 0) ? 0.125f : 1.0f;  // 1/sqrt(64) folded into Q
    half_t* out = (which == 0) ? Qh : Kh;
    #pragma unroll
    for (int t = 0; t < 4; ++t)
      #pragma unroll
      for (int j = 0; j < 8; ++j) {
        const int m = mt * 64 + wave * 16 + rsel + j;
        out[(((size_t)(b * kH + hh)) * kM + m) * kDH + t * 16 + col] =
            (half_t)(acc[t][j] * qscale);
      }
  } else {
    // --- V: transpose through LDS, write [b][h][dh][m] coalesced ---
    #pragma unroll
    for (int t = 0; t < 4; ++t)
      #pragma unroll
      for (int j = 0; j < 8; ++j)
        Vtr[(t * 16 + col) * 72 + wave * 16 + rsel + j] = (half_t)acc[t][j];
    __syncthreads();
    const int dh = tid >> 1;
    const int m0 = (tid & 1) * 32;
    half_t* dst = Vth + (((size_t)(b * kH + hh)) * kDH + dh) * kM + mt * 64 + m0;
    const v8h* srow = (const v8h*)&Vtr[dh * 72 + m0];
    #pragma unroll
    for (int i = 0; i < 4; ++i) ((v8h*)dst)[i] = srow[i];
  }
}

// ---------------------------------------------------------------------------
// Kernel B: flash attention. K/V tiles are staged global->LDS with
// double-buffered GLOBAL_LOAD_ASYNC_TO_LDS_B128 (ASYNCcnt), overlapping the
// next tile's fetch with the current tile's WMMA + online softmax.
// Block = 128 threads (4 waves); each wave owns 16 query rows.
// ---------------------------------------------------------------------------
__global__ __launch_bounds__(128)
void flash_attn_kernel(const half_t* __restrict__ Qg, const half_t* __restrict__ Kg,
                       const half_t* __restrict__ Vtg, const int* __restrict__ mask,
                       float* __restrict__ out)
{
  constexpr int kNT       = kM / 64;       // 16 key tiles
  constexpr int kTileB    = 64 * 72 * 2;   // bytes per (padded) LDS tile

  const int mt = blockIdx.x;
  const int hh = blockIdx.y;
  const int b  = blockIdx.z;
  const int tid  = threadIdx.x;
  const int lane = tid & 31;
  const int wave = tid >> 5;
  const int col  = lane & 15;
  const int rsel = (lane >> 4) * 8;
  const int koff = (lane < 16) ? 0 : 8;

  // Double-buffered K [key][72] and V [dh][72] tiles + per-wave P scratch.
  __shared__ half_t Kbuf[2][64 * 72];
  __shared__ half_t Vbuf[2][64 * 72];
  __shared__ half_t pbuf[4][16 * 72];
  half_t* pw = pbuf[wave];

  const size_t bh = (size_t)(b * kH + hh);
  const half_t* Qbh = Qg  + bh * kM * kDH;
  const half_t* Kbh = Kg  + bh * kM * kDH;   // [m][dh]
  const half_t* Vbh = Vtg + bh * kDH * kM;   // [dh][m]
  const int* mk = mask + (size_t)b * kM;

  // Per-lane async-copy geometry: each wave stages 16 rows of each tile,
  // 4 instructions x (32 lanes x 16B) = 4 rows per instruction.
  const unsigned kBase = (unsigned)(uintptr_t)(&Kbuf[0][0]);
  const unsigned vBase = (unsigned)(uintptr_t)(&Vbuf[0][0]);
  const int chunkCol = (lane & 7) * 16;           // byte offset within 128B row
  const int rowBase  = wave * 16 + (lane >> 3);   // + i*4

  auto stage_tiles = [&](int buf, int kb) {
    #pragma unroll
    for (int i = 0; i < 4; ++i) {
      const int row = rowBase + i * 4;
      // K: global [kb+row][dh] row = 128B
      async_copy_b128(kBase + (unsigned)(buf * kTileB + row * 144 + chunkCol),
                      (unsigned)((kb + row) * 128 + chunkCol), Kbh);
      // V: global [dh=row][kb..kb+63] segment = 128B (dh row stride = 2048B)
      async_copy_b128(vBase + (unsigned)(buf * kTileB + row * 144 + chunkCol),
                      (unsigned)(row * 2048 + kb * 2 + chunkCol), Vbh);
    }
  };

  // Q A-fragments straight from global ([m][dh] row-major, 16B chunks).
  const int qrow = mt * 64 + wave * 16 + col;
  Frag qa0, qa1;
  qa0.h[0] = *(const v8h*)&Qbh[(size_t)qrow * kDH + koff];
  qa0.h[1] = *(const v8h*)&Qbh[(size_t)qrow * kDH + koff + 16];
  qa1.h[0] = *(const v8h*)&Qbh[(size_t)qrow * kDH + 32 + koff];
  qa1.h[1] = *(const v8h*)&Qbh[(size_t)qrow * kDH + 48 + koff];

  float mrow[8], lrow[8];
  v8f oacc[4] = {};
  #pragma unroll
  for (int j = 0; j < 8; ++j) { mrow[j] = -1e30f; lrow[j] = 0.0f; }

  stage_tiles(0, 0);  // prologue: fill buffer 0 (8 async ops in flight)

  int cur = 0;
  for (int nt = 0; nt < kNT; ++nt) {
    const int kb = nt * 64;
    if (nt + 1 < kNT) {
      // Issue next tile (8 more async ops), then retire the 8 older ones:
      // async loads complete in order, so cnt<=8 means current buffer ready.
      stage_tiles(cur ^ 1, kb + 64);
      asm volatile("s_wait_asynccnt 8" ::: "memory");
    } else {
      asm volatile("s_wait_asynccnt 0" ::: "memory");
    }
    __syncthreads();  // all waves' quarters of current buffer landed

    const half_t* Kl = Kbuf[cur];
    const half_t* Vl = Vbuf[cur];

    // --- S = Q * K^T from LDS (conflict-free ds_load_b128, stride 144B) ---
    v8f s[4];
    #pragma unroll
    for (int t = 0; t < 4; ++t) {
      const half_t* krow = &Kl[(t * 16 + col) * 72];
      Frag bk0, bk1;
      bk0.h[0] = *(const v8h*)&krow[koff];
      bk0.h[1] = *(const v8h*)&krow[koff + 16];
      bk1.h[0] = *(const v8h*)&krow[32 + koff];
      bk1.h[1] = *(const v8h*)&krow[48 + koff];
      v8f z = {};
      z    = wmma_f16(qa0, bk0, z);
      s[t] = wmma_f16(qa1, bk1, z);
    }
    // --- key-side attention mask ---
    #pragma unroll
    for (int t = 0; t < 4; ++t) {
      if (mk[kb + t * 16 + col] == 0) {
        #pragma unroll
        for (int j = 0; j < 8; ++j) s[t][j] = -1e30f;
      }
    }
    // --- online softmax: row max (xor-shuffle within 16-lane halves) ---
    float rmax[8];
    #pragma unroll
    for (int j = 0; j < 8; ++j)
      rmax[j] = fmaxf(fmaxf(s[0][j], s[1][j]), fmaxf(s[2][j], s[3][j]));
    #pragma unroll
    for (int off = 8; off >= 1; off >>= 1)
      #pragma unroll
      for (int j = 0; j < 8; ++j)
        rmax[j] = fmaxf(rmax[j], __shfl_xor(rmax[j], off, 32));
    float alpha[8];
    #pragma unroll
    for (int j = 0; j < 8; ++j) {
      const float mn = fmaxf(mrow[j], rmax[j]);
      alpha[j] = __expf(mrow[j] - mn);
      mrow[j]  = mn;
    }
    // --- p = exp(s - m); row sum ---
    float rsum[8];
    #pragma unroll
    for (int j = 0; j < 8; ++j) rsum[j] = 0.0f;
    #pragma unroll
    for (int t = 0; t < 4; ++t)
      #pragma unroll
      for (int j = 0; j < 8; ++j) {
        const float p = __expf(s[t][j] - mrow[j]);
        s[t][j] = p;
        rsum[j] += p;
      }
    #pragma unroll
    for (int off = 8; off >= 1; off >>= 1)
      #pragma unroll
      for (int j = 0; j < 8; ++j)
        rsum[j] += __shfl_xor(rsum[j], off, 32);
    #pragma unroll
    for (int j = 0; j < 8; ++j) lrow[j] = lrow[j] * alpha[j] + rsum[j];
    #pragma unroll
    for (int t = 0; t < 4; ++t)
      #pragma unroll
      for (int j = 0; j < 8; ++j) oacc[t][j] *= alpha[j];

    // --- P: C-layout regs -> per-wave LDS row-major [qrow16][key64] ---
    #pragma unroll
    for (int t = 0; t < 4; ++t) {
      const int kk = t * 16 + col;
      #pragma unroll
      for (int j = 0; j < 8; ++j)
        pw[(rsel + j) * 72 + kk] = (half_t)s[t][j];
    }
    asm volatile("s_wait_dscnt 0" ::: "memory");  // same-wave DS RAW fence

    Frag pa0, pa1;
    pa0.h[0] = *(const v8h*)&pw[col * 72 + koff];
    pa0.h[1] = *(const v8h*)&pw[col * 72 + koff + 16];
    pa1.h[0] = *(const v8h*)&pw[col * 72 + 32 + koff];
    pa1.h[1] = *(const v8h*)&pw[col * 72 + 48 + koff];

    // --- O += P * V (V B-fragments from LDS [dh][key]) ---
    #pragma unroll
    for (int t = 0; t < 4; ++t) {
      const half_t* vrow = &Vl[(t * 16 + col) * 72];
      Frag bv0, bv1;
      bv0.h[0] = *(const v8h*)&vrow[koff];
      bv0.h[1] = *(const v8h*)&vrow[koff + 16];
      bv1.h[0] = *(const v8h*)&vrow[32 + koff];
      bv1.h[1] = *(const v8h*)&vrow[48 + koff];
      oacc[t] = wmma_f16(pa0, bv0, oacc[t]);
      oacc[t] = wmma_f16(pa1, bv1, oacc[t]);
    }
    __syncthreads();  // all waves done reading cur before it is overwritten
    cur ^= 1;
  }

  // --- normalize and store fp32 [b][h][m][dh] ---
  #pragma unroll
  for (int t = 0; t < 4; ++t)
    #pragma unroll
    for (int j = 0; j < 8; ++j) {
      const int m = mt * 64 + wave * 16 + rsel + j;
      out[(bh * kM + m) * kDH + t * 16 + col] = oacc[t][j] / lrow[j];
    }
}

// ---------------------------------------------------------------------------
extern "C" void kernel_launch(void* const* d_in, const int* in_sizes, int n_in,
                              void* d_out, int out_size, void* d_ws, size_t ws_size,
                              hipStream_t stream) {
  const float* Pq = (const float*)d_in[0];
  const float* Pk = (const float*)d_in[1];
  const float* Pv = (const float*)d_in[2];
  const float* Vq = (const float*)d_in[3];
  const float* Vk = (const float*)d_in[4];
  const float* Vv = (const float*)d_in[5];
  const float* bq = (const float*)d_in[6];
  const float* bk = (const float*)d_in[7];
  const float* bv = (const float*)d_in[8];
  const int* mask = (const int*)d_in[9];
  const int* pos  = (const int*)d_in[10];
  float* out = (float*)d_out;

  const int B = in_sizes[0] / (kM * kR);

  // Workspace: Q, K ([B][H][M][DH]) and V transposed ([B][H][DH][M]), f16.
  const size_t perT = (size_t)B * kH * kM * kDH;
  half_t* Qh  = (half_t*)d_ws;
  half_t* Kh  = Qh + perT;
  half_t* Vth = Kh + perT;

  dim3 blk(128);
  dim3 grdA(kM / 64, kH, B * 3);
  hipLaunchKernelGGL(svd_proj_rope_kernel, grdA, blk, 0, stream,
                     Pq, Pk, Pv, Vq, Vk, Vv, bq, bk, bv, pos, Qh, Kh, Vth);

  dim3 grdB(kM / 64, kH, B);
  hipLaunchKernelGGL(flash_attn_kernel, grdB, blk, 0, stream,
                     Qh, Kh, Vth, mask, out);
}
